// EquivariantLongRangeAttention_1821066133971
// MI455X (gfx1250) — matrix operand order repaired and checked by reading, not compiled, not run on hardware
//
#include <hip/hip_runtime.h>
#include <math.h>

// Problem constants (from reference)
#define DIM    128
#define NHEAD  4
#define NGRAPH 48
#define ATTN_SCALE 0.1767766952966369f   // (DIM/NHEAD)^-0.5 = 32^-0.5

#define VROW 40                          // padded LDS row (bf16) for a 32-key slab row

typedef __attribute__((ext_vector_type(16))) __bf16 v16bf;
typedef __attribute__((ext_vector_type(8)))  __bf16 v8bf;
typedef __attribute__((ext_vector_type(8)))  float  v8f;
typedef __attribute__((ext_vector_type(4)))  float  v4f;
typedef __attribute__((ext_vector_type(4)))  int    v4i;

// ---------------------------------------------------------------------------
// gfx1250 async global->LDS copy (ASYNCcnt-tracked). Guarded so the file
// compiles on toolchains without the builtin (falls back to register bounce).
// Builtin signature (probe-confirmed by diagnostic): (int4 AS1*, int4 AS3*,
// imm offset, imm cpol).
// ---------------------------------------------------------------------------
#if defined(__has_builtin)
#if __has_builtin(__builtin_amdgcn_global_load_async_to_lds_b128)
#define HAVE_ASYNC_LDS 1
#endif
#endif
#ifndef HAVE_ASYNC_LDS
#define HAVE_ASYNC_LDS 0
#endif

__device__ inline void async_copy16(const void* g, void* l) {
#if HAVE_ASYNC_LDS
  const v4i* gv = (const v4i*)g;   // generic
  v4i*       lv = (v4i*)l;         // generic (points into __shared__)
  __builtin_amdgcn_global_load_async_to_lds_b128(
      (__attribute__((address_space(1))) v4i*)gv,
      (__attribute__((address_space(3))) v4i*)lv, 0, 0);
#else
  *(v8bf*)l = *(const v8bf*)g;
#endif
}

__device__ inline void async_wait(bool pendingNext) {
#if HAVE_ASYNC_LDS
  if (pendingNext) asm volatile("s_wait_asynccnt 0x2" ::: "memory");
  else             asm volatile("s_wait_asynccnt 0x0" ::: "memory");
#endif
}

// ---------------------------------------------------------------------------
// Fragment loaders (A/B use this file's contiguous k order, not the ISA table order; C/D per CDNA5 ISA)
//
// A (16x32 bf16, M x K): lane = m + 16*halfA
//   elems 0..15 -> K = 16*halfA + e         (contiguous run of 16)
//
// B (32x16 bf16, K x N): lane = n + 16*half
//   elems 0..15 -> K = 16*half + e          (contiguous run of 16)
//
// C/D (16x16 f32): lane = n + 16*half, elem r -> row m = r + 8*half
// ---------------------------------------------------------------------------

__device__ inline v16bf load_afrag_f32(const float* __restrict__ rowp, int k0, int half) {
  v16bf a;
  const float* p0 = rowp + k0 + 16 * half;
  const float* p1 = rowp + k0 + 16 * half + 8;
#pragma unroll
  for (int e = 0; e < 8; ++e) { a[e] = (__bf16)p0[e]; a[8 + e] = (__bf16)p1[e]; }
  return a;
}

__device__ inline v16bf load_bfrag(const __bf16* __restrict__ T, int ld, int col,
                                   int kb, int half) {
  const __bf16* p = T + (size_t)col * ld + kb + 16 * half;
  v8bf lo = *(const v8bf*)(p);
  v8bf hi = *(const v8bf*)(p + 8);
  v16bf b;
#pragma unroll
  for (int e = 0; e < 8; ++e) { b[e] = lo[e]; b[8 + e] = hi[e]; }
  return b;
}

// B-fragment from the staged LDS slab (padded row = VROW bf16)
__device__ inline v16bf load_bfrag_lds(const __bf16* lbuf, int col, int half) {
  const __bf16* p = lbuf + col * VROW + 16 * half;
  v8bf lo = *(const v8bf*)(p);
  v8bf hi = *(const v8bf*)(p + 8);
  v16bf b;
#pragma unroll
  for (int e = 0; e < 8; ++e) { b[e] = lo[e]; b[8 + e] = hi[e]; }
  return b;
}

// ---------------------------------------------------------------------------
// Kernel 1: graph boundary offsets via binary search on sorted int64 batch
// ---------------------------------------------------------------------------
__global__ void compute_starts_kernel(const long long* __restrict__ batch, int n,
                                      int* __restrict__ starts) {
  int g = threadIdx.x;
  if (g > NGRAPH) return;
  int lo = 0, hi = n;
  while (lo < hi) {
    int mid = (lo + hi) >> 1;
    if (batch[mid] < (long long)g) lo = mid + 1; else hi = mid;
  }
  starts[g] = lo;
}

// ---------------------------------------------------------------------------
// Kernel 2: WvT[d][k] = bf16(Wv[k][d])  (so B-fragments load contiguously)
// ---------------------------------------------------------------------------
__global__ void prep_wvt_kernel(const float* __restrict__ Wv, __bf16* __restrict__ WvT) {
  int idx = blockIdx.x * blockDim.x + threadIdx.x;
  if (idx >= DIM * DIM) return;
  int d = idx >> 7, kk = idx & (DIM - 1);
  WvT[(size_t)d * DIM + kk] = (__bf16)Wv[(size_t)kk * DIM + d];
}

// ---------------------------------------------------------------------------
// Kernel 3: zero the vT column pad [n, ldv) so 32-wide key tiles can overread
// ---------------------------------------------------------------------------
__global__ void pad_vt_kernel(__bf16* __restrict__ vT, int ldv, int n) {
  int idx = blockIdx.x * blockDim.x + threadIdx.x;  // DIM * 64 threads
  int d = idx >> 6, c = idx & 63;
  if (d >= DIM) return;
  if (n + c < ldv) vT[(size_t)d * ldv + n + c] = (__bf16)0.0f;
}

// ---------------------------------------------------------------------------
// Kernel 4: q/k projections (H=4 -> VALU, weights staged in LDS)
// ---------------------------------------------------------------------------
__global__ void proj_qk_kernel(const float* __restrict__ hs,
                               const float* __restrict__ Wq, const float* __restrict__ bq,
                               const float* __restrict__ Wk, const float* __restrict__ bk,
                               float* __restrict__ q, float* __restrict__ k, int n) {
  __shared__ float wql[DIM * NHEAD];
  __shared__ float wkl[DIM * NHEAD];
  for (int i = threadIdx.x; i < DIM * NHEAD; i += blockDim.x) { wql[i] = Wq[i]; wkl[i] = Wk[i]; }
  __syncthreads();
  int node = blockIdx.x * blockDim.x + threadIdx.x;
  if (node >= n) return;
  const float* row = hs + (size_t)node * DIM;
  float aq[NHEAD] = {0.f, 0.f, 0.f, 0.f};
  float ak[NHEAD] = {0.f, 0.f, 0.f, 0.f};
  for (int c = 0; c < DIM; ++c) {
    float hv = row[c];
#pragma unroll
    for (int hh = 0; hh < NHEAD; ++hh) {
      aq[hh] = fmaf(hv, wql[c * NHEAD + hh], aq[hh]);
      ak[hh] = fmaf(hv, wkl[c * NHEAD + hh], ak[hh]);
    }
  }
#pragma unroll
  for (int hh = 0; hh < NHEAD; ++hh) {
    q[(size_t)node * NHEAD + hh] = aq[hh] + bq[hh];
    k[(size_t)node * NHEAD + hh] = ak[hh] + bk[hh];
  }
}

// ---------------------------------------------------------------------------
// Kernel 5: v = h @ Wv + bv via WMMA bf16 -> stored transposed bf16 (vT[d][node])
// One 16x16 output tile per wave: rowTile = blockIdx.x, colTile = wave (8 waves)
// ---------------------------------------------------------------------------
__global__ void proj_v_kernel(const float* __restrict__ h, const __bf16* __restrict__ WvT,
                              const float* __restrict__ bv, __bf16* __restrict__ vT,
                              int ldv, int n) {
  int lane = threadIdx.x & 31, wave = threadIdx.x >> 5;
  int row0 = blockIdx.x * 16;
  int d0 = wave * 16;
  if (row0 >= n) return;
  int m = lane & 15, half = lane >> 4;
  int arow_idx = row0 + m; if (arow_idx >= n) arow_idx = n - 1;   // clamp tail
  const float* arow = h + (size_t)arow_idx * DIM;
  v8f acc = {};
#pragma unroll
  for (int kt = 0; kt < DIM / 32; ++kt) {
    v16bf a = load_afrag_f32(arow, kt * 32, half);
    v16bf b = load_bfrag(WvT, DIM, d0 + m, kt * 32, half);
    acc = __builtin_amdgcn_wmma_f32_16x16x32_bf16(false, a, false, b, (short)0, acc,
                                                  false, false);
  }
  int col = d0 + m;
  float bias = bv[col];
#pragma unroll
  for (int r = 0; r < 8; ++r) {
    int row = row0 + r + 8 * half;
    if (row < n) vT[(size_t)col * ldv + row] = (__bf16)(acc[r] + bias);
  }
}

// ---------------------------------------------------------------------------
// Kernel 6: per-graph flash attention. 1 block/graph, 8 waves, 1 16-row tile
// per wave per pass. All waves march over the same aligned 32-key blocks; the
// 32x128 V slab is staged into LDS (double-buffered) with gfx1250 async
// global->LDS copies so staging of block i+1 overlaps compute of block i.
// P goes through per-wave LDS to convert C-layout -> A-layout (bf16), then
// 8 WMMAs per 32 keys accumulate the 16x128 output tile.
// ---------------------------------------------------------------------------
__global__ void attn_kernel(const float* __restrict__ q, const float* __restrict__ k,
                            const __bf16* __restrict__ vT, int ldv,
                            const int* __restrict__ starts, float* __restrict__ out) {
  __shared__ __bf16 plds[8][16 * 32];          // per-wave P tile (16 rows x 32 keys)
  __shared__ __bf16 vlds[2][DIM * VROW];       // double-buffered V slab (32 keys)

  int g = blockIdx.x;
  int s0 = starts[g], e0 = starts[g + 1];
  if (e0 <= s0) return;                        // empty graph (uniform exit)
  int tid = threadIdx.x;
  int lane = tid & 31, wave = tid >> 5;
  int n = lane & 15, half = lane >> 4;
  __bf16* pbuf = plds[wave];
  int ntiles = (e0 - s0 + 15) >> 4;
  int passes = (ntiles + 7) >> 3;
  int kb0 = s0 & ~31;                          // aligned key blocks

  for (int pass = 0; pass < passes; ++pass) {
    int tile = pass * 8 + wave;                // wave-uniform
    bool active = tile < ntiles;
    int row0 = s0 + tile * 16;

    v4f qr[8];
#pragma unroll
    for (int r = 0; r < 8; ++r) {
      int gr = row0 + r + 8 * half;
      v4f qv = {};
      if (active && gr < e0) qv = *(const v4f*)(q + (size_t)gr * NHEAD);
      qr[r] = qv;
    }

    v8f acc[8];
#pragma unroll
    for (int t = 0; t < 8; ++t) acc[t] = (v8f){};
    float rmax[8], rsum[8];
#pragma unroll
    for (int r = 0; r < 8; ++r) { rmax[r] = -__builtin_inff(); rsum[r] = 0.f; }

    // Prologue: stage first V slab (2 async b128 per thread -> 8KB slab)
#pragma unroll
    for (int c = tid; c < DIM * 4; c += 256) {
      int d = c >> 2, part = c & 3;
      async_copy16(vT + (size_t)d * ldv + kb0 + part * 8, &vlds[0][d * VROW + part * 8]);
    }

    int parity = 0;
    for (int kb = kb0; kb < e0; kb += 32) {
      bool hasNext = (kb + 32) < e0;
      if (hasNext) {                           // prefetch next slab into other buffer
        int kn = kb + 32;
#pragma unroll
        for (int c = tid; c < DIM * 4; c += 256) {
          int d = c >> 2, part = c & 3;
          async_copy16(vT + (size_t)d * ldv + kn + part * 8,
                       &vlds[1 - parity][d * VROW + part * 8]);
        }
      }
      async_wait(hasNext);                     // current slab's copies complete
      __syncthreads();                         // visible to all waves

      if (active) {
        int j0 = kb + n, j1 = kb + 16 + n;
        bool v0 = (j0 >= s0) && (j0 < e0);
        bool v1 = (j1 >= s0) && (j1 < e0);
        v4f k0v = *(const v4f*)(k + (size_t)(v0 ? j0 : s0) * NHEAD);
        v4f k1v = *(const v4f*)(k + (size_t)(v1 ? j1 : s0) * NHEAD);

        float p0[8], p1[8];
#pragma unroll
        for (int r = 0; r < 8; ++r) {
          float sa = (qr[r][0]*k0v[0] + qr[r][1]*k0v[1] + qr[r][2]*k0v[2] + qr[r][3]*k0v[3]) * ATTN_SCALE;
          float sb = (qr[r][0]*k1v[0] + qr[r][1]*k1v[1] + qr[r][2]*k1v[2] + qr[r][3]*k1v[3]) * ATTN_SCALE;
          sa = v0 ? sa : -__builtin_inff();
          sb = v1 ? sb : -__builtin_inff();
          float loc = fmaxf(sa, sb);           // row max over 16-lane half-group
#pragma unroll
          for (int msk = 1; msk < 16; msk <<= 1) loc = fmaxf(loc, __shfl_xor(loc, msk, 32));
          float mn = fmaxf(rmax[r], loc);
          float pa = __expf(sa - mn);
          float pb = __expf(sb - mn);
          float corr = __expf(rmax[r] - mn);   // first iter: exp(-inf)=0
          rmax[r] = mn;
          float ls = pa + pb;
#pragma unroll
          for (int msk = 1; msk < 16; msk <<= 1) ls += __shfl_xor(ls, msk, 32);
          rsum[r] = rsum[r] * corr + ls;
          p0[r] = pa; p1[r] = pb;
#pragma unroll
          for (int t = 0; t < 8; ++t) acc[t][r] *= corr;
        }

        // C-layout P -> LDS (row-major 16x32 bf16), reread as A-fragment
#pragma unroll
        for (int r = 0; r < 8; ++r) {
          int mrow = r + 8 * half;
          pbuf[mrow * 32 + n]      = (__bf16)p0[r];
          pbuf[mrow * 32 + 16 + n] = (__bf16)p1[r];
        }
        v16bf a;
        {
          const __bf16* pA0 = &pbuf[n * 32 + 16 * half];
          const __bf16* pA1 = &pbuf[n * 32 + 16 * half + 8];
#pragma unroll
          for (int e = 0; e < 8; ++e) { a[e] = pA0[e]; a[8 + e] = pA1[e]; }
        }

        const __bf16* vbuf = vlds[parity];
#pragma unroll
        for (int t = 0; t < 8; ++t) {
          v16bf b = load_bfrag_lds(vbuf, t * 16 + n, half);
          acc[t] = __builtin_amdgcn_wmma_f32_16x16x32_bf16(false, a, false, b, (short)0,
                                                           acc[t], false, false);
        }
      }

      __syncthreads();                         // done reading slab before re-stage
      parity ^= 1;
    }

    // epilogue: divide by row sums, write f32 output
    if (active) {
#pragma unroll
      for (int r = 0; r < 8; ++r) {
        int gr = row0 + r + 8 * half;
        if (gr < e0) {
          float inv = 1.0f / rsum[r];
#pragma unroll
          for (int t = 0; t < 8; ++t)
            out[(size_t)gr * DIM + t * 16 + n] = acc[t][r] * inv;
        }
      }
    }
  }
}

// ---------------------------------------------------------------------------
extern "C" void kernel_launch(void* const* d_in, const int* in_sizes, int n_in,
                              void* d_out, int out_size, void* d_ws, size_t ws_size,
                              hipStream_t stream) {
  const float*     h     = (const float*)d_in[0];
  const float*     hs    = (const float*)d_in[1];
  const long long* batch = (const long long*)d_in[2];
  const float*     Wq    = (const float*)d_in[3];
  const float*     bq    = (const float*)d_in[4];
  const float*     Wk    = (const float*)d_in[5];
  const float*     bk    = (const float*)d_in[6];
  const float*     Wv    = (const float*)d_in[7];
  const float*     bv    = (const float*)d_in[8];
  float* out = (float*)d_out;

  int n   = in_sizes[0] / DIM;
  int ldv = (n + 32 + 15) & ~15;            // padded, 16-element aligned lead dim

  // Workspace carve-up (all offsets stay 256B aligned for n multiple of 16)
  char* ws = (char*)d_ws;
  float* qbuf = (float*)ws;  ws += (size_t)n * NHEAD * sizeof(float);
  float* kbuf = (float*)ws;  ws += (size_t)n * NHEAD * sizeof(float);
  int* starts = (int*)ws;    ws += 256;
  __bf16* WvT = (__bf16*)ws; ws += (size_t)DIM * DIM * sizeof(__bf16);
  __bf16* vT  = (__bf16*)ws; // DIM * ldv bf16

  compute_starts_kernel<<<1, 64, 0, stream>>>(batch, n, starts);
  prep_wvt_kernel<<<(DIM * DIM + 255) / 256, 256, 0, stream>>>(Wv, WvT);
  pad_vt_kernel<<<(DIM * 64 + 255) / 256, 256, 0, stream>>>(vT, ldv, n);
  proj_qk_kernel<<<(n + 255) / 256, 256, 0, stream>>>(hs, Wq, bq, Wk, bk, qbuf, kbuf, n);
  proj_v_kernel<<<(n + 15) / 16, 256, 0, stream>>>(h, WvT, bv, vT, ldv, n);
  attn_kernel<<<NGRAPH, 256, 0, stream>>>(qbuf, kbuf, vT, ldv, starts, out);
}
